// StochasticDecoder_75634374082628
// MI455X (gfx1250) — matrix operand order repaired and checked by reading, not compiled, not run on hardware
//
#include <hip/hip_runtime.h>
#include <hip/hip_bf16.h>
#include <stdint.h>

#define E 128
#define V 32000
#define T 16
#define BATCH 4096

#define NSPLIT 8            // column splits across blocks
#define TILES_PER_SPLIT 250 // 2000 / 8
#define TILES_PER_CHUNK 10
#define NCHUNK 25           // 250 / 10
#define CHUNK_COLS (TILES_PER_CHUNK * 16)       // 160 columns
#define CHUNK_ELEMS (CHUNK_COLS * E)            // 20480 bf16 = 40960 B

typedef __attribute__((ext_vector_type(16))) __bf16 v16bf;
typedef __attribute__((ext_vector_type(8)))  float  v8f;
typedef __attribute__((ext_vector_type(4)))  float  vf4;
typedef __attribute__((ext_vector_type(4)))  unsigned int u32x4;
typedef __attribute__((ext_vector_type(8)))  int    i32x8;
typedef __attribute__((ext_vector_type(4)))  int    i32x4;

union V16A { v16bf v; vf4 f[2]; };

__device__ __forceinline__ uint32_t pcg_hash(uint32_t x) {
  x = x * 747796405u + 2891336453u;
  x = ((x >> ((x >> 28u) + 4u)) ^ x) * 277803737u;
  return (x >> 22u) ^ x;
}

// TDM: stage one contiguous 40960-byte W_out chunk into LDS.
// D# per CDNA5 ISA ch.8: group0 = {count=1, lds_addr, global_addr, type=2},
// group1 = {data_size=2B, dim0=tile0=20480, dim1=tile1=1, stride=20480}.
// This toolchain exposes the 6-arg builtin (extra i32x8 group before cpol).
__device__ __forceinline__ void tdm_load_chunk(const __bf16* gp, uint32_t lds_off) {
  uint64_t ga = (uint64_t)(uintptr_t)gp;
  u32x4 g0 = {1u,                                  // count=1, is_restore=0, gather off
              lds_off,                              // [63:32] lds_addr
              (uint32_t)ga,                         // [95:64] global_addr lo
              (uint32_t)((ga >> 32) & 0x01FFFFFFu) | 0x80000000u}; // addr hi | type=2
  i32x8 g1 = {0x00010000,                           // data_size=2 bytes
              0x50000000,                           // tensor_dim0[15:0]=20480 << 16
              0x00010000,                           // tensor_dim0 hi=0, tensor_dim1=1
              0x50000000,                           // tile_dim0=20480 << 16
              1,                                    // tile_dim1=1, tile_dim2=0
              20480,                                // tensor_dim0_stride lo
              0x50000000,                           // stride0 hi=0, stride1 lo=20480
              0};
  i32x4 z4 = {0, 0, 0, 0};
  i32x8 z8 = {0, 0, 0, 0, 0, 0, 0, 0};
  __builtin_amdgcn_tensor_load_to_lds(g0, g1, z4, z4, z8, 0);
}

// ---------------------------------------------------------------------------
// Init / convert
// ---------------------------------------------------------------------------
__global__ void init_kernel(const float* __restrict__ x, float* __restrict__ state,
                            __bf16* __restrict__ state_b, int* __restrict__ last_tok,
                            int* __restrict__ alive, int* __restrict__ n_outer,
                            float* __restrict__ ent_rows) {
  int i = blockIdx.x * blockDim.x + threadIdx.x;
  if (i < BATCH * E) {
    float v = x[i];
    state[i] = v;
    state_b[i] = (__bf16)v;
  }
  if (i < BATCH) {
    last_tok[i] = 0;
    alive[i] = 1;
    n_outer[i] = T;
    ent_rows[i] = 0.0f;
  }
}

__global__ void convert_wout(const float* __restrict__ w, __bf16* __restrict__ wb) {
  int i = blockIdx.x * blockDim.x + threadIdx.x;
  if (i < V * E) wb[i] = (__bf16)w[i];
}

// ---------------------------------------------------------------------------
// GRU cell: one row per 128-thread block.
// ---------------------------------------------------------------------------
__launch_bounds__(128)
__global__ void gru_step_kernel(float* __restrict__ state, __bf16* __restrict__ state_b,
                                const int* __restrict__ last_tok, const int* __restrict__ alive,
                                const float* __restrict__ emb_table,
                                const float* __restrict__ W_ih, const float* __restrict__ W_hh,
                                const float* __restrict__ b_ih, const float* __restrict__ b_hh) {
  __shared__ vf4 es[32];
  __shared__ vf4 hs[32];
  const int row = blockIdx.x;
  const int j = threadIdx.x;
  const int tok = last_tok[row];

  ((float*)es)[j] = emb_table[(size_t)tok * E + j];
  ((float*)hs)[j] = state[(size_t)row * E + j];
  __syncthreads();

  float gir = b_ih[j], giz = b_ih[j + E], gin = b_ih[j + 2 * E];
  float ghr = b_hh[j], ghz = b_hh[j + E], ghn = b_hh[j + 2 * E];

  const vf4* wir = (const vf4*)(W_ih + (size_t)j * E);
  const vf4* wiz = (const vf4*)(W_ih + (size_t)(j + E) * E);
  const vf4* win = (const vf4*)(W_ih + (size_t)(j + 2 * E) * E);
  const vf4* whr = (const vf4*)(W_hh + (size_t)j * E);
  const vf4* whz = (const vf4*)(W_hh + (size_t)(j + E) * E);
  const vf4* whn = (const vf4*)(W_hh + (size_t)(j + 2 * E) * E);

#pragma unroll 8
  for (int k = 0; k < 32; ++k) {
    vf4 e4 = es[k], h4 = hs[k], a;
    a = wir[k]; gir += e4.x * a.x + e4.y * a.y + e4.z * a.z + e4.w * a.w;
    a = wiz[k]; giz += e4.x * a.x + e4.y * a.y + e4.z * a.z + e4.w * a.w;
    a = win[k]; gin += e4.x * a.x + e4.y * a.y + e4.z * a.z + e4.w * a.w;
    a = whr[k]; ghr += h4.x * a.x + h4.y * a.y + h4.z * a.z + h4.w * a.w;
    a = whz[k]; ghz += h4.x * a.x + h4.y * a.y + h4.z * a.z + h4.w * a.w;
    a = whn[k]; ghn += h4.x * a.x + h4.y * a.y + h4.z * a.z + h4.w * a.w;
  }

  float rg = 1.0f / (1.0f + __expf(-(gir + ghr)));
  float zg = 1.0f / (1.0f + __expf(-(giz + ghz)));
  float ng = tanhf(gin + rg * ghn);
  float h_old = ((float*)hs)[j];
  float h_new = (1.0f - zg) * ng + zg * h_old;
  if (!alive[row]) h_new = h_old;
  state[(size_t)row * E + j] = h_new;
  state_b[(size_t)row * E + j] = (__bf16)h_new;
}

// ---------------------------------------------------------------------------
// Phase 1: fused logits GEMM (bf16 WMMA) + softmax sums + Gumbel argmax.
// Block = 128 rows x 250 column tiles; 8 waves share TDM-staged W_out chunks
// in LDS (double-buffered, 40 KB each). Branch-free epilogue (|logit| << 80
// for this model, so no running max is needed -> partials are plain sums).
// B fragments for all 4 k-blocks are loaded into distinct registers before
// the WMMA chain so the ds_loads clause and overlap the matrix pipe.
// ---------------------------------------------------------------------------
__launch_bounds__(256)
__global__ void logits_partial_kernel(const __bf16* __restrict__ stateb,
                                      const __bf16* __restrict__ woutb,
                                      const float* __restrict__ b_out,
                                      float* __restrict__ pZ, float* __restrict__ pSL,
                                      float* __restrict__ pBK, int* __restrict__ pBI,
                                      int t) {
  __shared__ __align__(32) __bf16 sB[2][CHUNK_ELEMS];

  const int tid = threadIdx.x;
  const int wave = tid >> 5;
  const int lane = tid & 31;
  const int l15 = lane & 15;
  const int g = lane >> 4;
  const int mblk = blockIdx.x >> 3;       // 0..31  (row block of 128)
  const int s = blockIdx.x & 7;           // 0..7   (column split)
  const int mbase = mblk * 128 + wave * 16;
  const int tile0 = s * TILES_PER_SPLIT;

  // A tile: this wave's 16 rows x 128 K in registers (contiguous k = 32*kb + 16g + s, same order as B).
  const __bf16* arow = stateb + (size_t)(mbase + l15) * E + 16 * g;
  V16A A[4];
#pragma unroll
  for (int kb = 0; kb < 4; ++kb) {
    A[kb].f[0] = *(const vf4*)(arow + 32 * kb);
    A[kb].f[1] = *(const vf4*)(arow + 32 * kb + 8);
  }

  float Zs[8], SL[8], bk[8];
  int bi[8];
  uint32_t rh[8];
#pragma unroll
  for (int r = 0; r < 8; ++r) {
    Zs[r] = 0.0f; SL[r] = 0.0f; bk[r] = -3.0e38f; bi[r] = 0;
    rh[r] = (uint32_t)(mbase + r + 8 * g) * 2654435761u ^ (uint32_t)t * 0x85EBCA6Bu;
  }

  if (wave == 0) {
    tdm_load_chunk(woutb + (size_t)tile0 * 16 * E, (uint32_t)(uintptr_t)&sB[0][0]);
  }

  for (int c = 0; c < NCHUNK; ++c) {
    if (wave == 0) {
      if (c + 1 < NCHUNK) {
        tdm_load_chunk(woutb + (size_t)(tile0 + (c + 1) * TILES_PER_CHUNK) * 16 * E,
                       (uint32_t)(uintptr_t)&sB[(c + 1) & 1][0]);
        __builtin_amdgcn_s_wait_tensorcnt(1);  // chunk c complete, c+1 in flight
      } else {
        __builtin_amdgcn_s_wait_tensorcnt(0);
      }
    }
    __syncthreads();  // chunk c visible; previous buffer fully consumed

    const __bf16* bb = &sB[c & 1][0];
#pragma unroll 2
    for (int j = 0; j < TILES_PER_CHUNK; ++j) {
      const int col = (tile0 + c * TILES_PER_CHUNK + j) * 16 + l15;
      // B tile from LDS: lane n = l15, elems = K 32*kb + 16g + (0..15) contiguous
      const __bf16* bw = bb + (size_t)(j * 16 + l15) * E + 16 * g;

      // Load all 4 k-block fragments into distinct registers first so the
      // DS loads issue as one clause and overlap the WMMA chain.
      const v16bf bm0 = *(const v16bf*)(bw);
      const v16bf bm1 = *(const v16bf*)(bw + 32);
      const v16bf bm2 = *(const v16bf*)(bw + 64);
      const v16bf bm3 = *(const v16bf*)(bw + 96);
      const float bo = b_out[col];

      v8f cacc = {0.f, 0.f, 0.f, 0.f, 0.f, 0.f, 0.f, 0.f};
      cacc = __builtin_amdgcn_wmma_f32_16x16x32_bf16(false, A[0].v, false, bm0,
                                                     (short)0, cacc, false, false);
      cacc = __builtin_amdgcn_wmma_f32_16x16x32_bf16(false, A[1].v, false, bm1,
                                                     (short)0, cacc, false, false);
      cacc = __builtin_amdgcn_wmma_f32_16x16x32_bf16(false, A[2].v, false, bm2,
                                                     (short)0, cacc, false, false);
      cacc = __builtin_amdgcn_wmma_f32_16x16x32_bf16(false, A[3].v, false, bm3,
                                                     (short)0, cacc, false, false);

#pragma unroll
      for (int r = 0; r < 8; ++r) {
        float logit = cacc[r] + bo;
        float e = __expf(logit);
        Zs[r] += e;
        SL[r] = fmaf(e, logit, SL[r]);
        uint32_t h = pcg_hash(rh[r] ^ (uint32_t)col * 0x9E3779B9u);
        float u = (float)(h >> 8) * (1.0f / 16777216.0f) + 2.98e-8f;
        float key = logit - __logf(-__logf(u));
        bool take = key > bk[r];
        bk[r] = take ? key : bk[r];
        bi[r] = take ? col : bi[r];
      }
    }
    __syncthreads();  // done reading buffer c&1 before it is re-staged
  }

  // Reduce across the 16 lanes of each half (plain sums + argmax).
#pragma unroll
  for (int mask = 1; mask <= 8; mask <<= 1) {
#pragma unroll
    for (int r = 0; r < 8; ++r) {
      Zs[r] += __shfl_xor(Zs[r], mask);
      SL[r] += __shfl_xor(SL[r], mask);
      float ob = __shfl_xor(bk[r], mask);
      int   oi = __shfl_xor(bi[r], mask);
      if (ob > bk[r]) { bk[r] = ob; bi[r] = oi; }
    }
  }

  if (l15 == 0) {
#pragma unroll
    for (int r = 0; r < 8; ++r) {
      int row = mbase + r + 8 * g;
      int idx = row * NSPLIT + s;
      pZ[idx] = Zs[r];
      pSL[idx] = SL[r];
      pBK[idx] = bk[r];
      pBI[idx] = bi[r];
    }
  }
}

// ---------------------------------------------------------------------------
// Phase 2: per-row combine of the 8 column-split partials; token/alive update.
// ---------------------------------------------------------------------------
__global__ void sample_finalize(const float* __restrict__ pZ, const float* __restrict__ pSL,
                                const float* __restrict__ pBK, const int* __restrict__ pBI,
                                int* __restrict__ alive, int* __restrict__ last_tok,
                                int* __restrict__ n_outer, float* __restrict__ ent_rows,
                                int* __restrict__ out_utt, int t) {
  int row = blockIdx.x * blockDim.x + threadIdx.x;
  if (row >= BATCH) return;
  float Z = 0.0f, S = 0.0f, k = -3.0e38f;
  int idx = 0;
#pragma unroll
  for (int s = 0; s < NSPLIT; ++s) {
    int p = row * NSPLIT + s;
    Z += pZ[p];
    S += pSL[p];
    float ob = pBK[p];
    if (ob > k) { k = ob; idx = pBI[p]; }
  }
  float ent = __logf(Z) - S / Z;  // H = logZ - E[l]  (max shift = 0)
  int alv = alive[row];
  int tok = alv ? idx : 0;
  if (alv) ent_rows[row] += ent;
  if (alv && tok == 0) n_outer[row] = t + 1;
  alive[row] = (alv && tok != 0) ? 1 : 0;
  last_tok[row] = tok;
  out_utt[row * T + t] = tok;
}

// ---------------------------------------------------------------------------
// Finalize outputs
// ---------------------------------------------------------------------------
__global__ void write_nouter(const int* __restrict__ n_outer, int* __restrict__ out_i) {
  int i = blockIdx.x * blockDim.x + threadIdx.x;
  if (i < BATCH) out_i[BATCH * T + i] = n_outer[i];
}

__global__ void entropy_reduce(const float* __restrict__ ent_rows, float* __restrict__ out_f) {
  __shared__ float red[256];
  int tid = threadIdx.x;
  float s = 0.0f;
  for (int i = tid; i < BATCH; i += 256) s += ent_rows[i];
  red[tid] = s;
  __syncthreads();
  for (int off = 128; off > 0; off >>= 1) {
    if (tid < off) red[tid] += red[tid + off];
    __syncthreads();
  }
  if (tid == 0) out_f[BATCH * T + BATCH] = red[0];
}

// ---------------------------------------------------------------------------
extern "C" void kernel_launch(void* const* d_in, const int* in_sizes, int n_in,
                              void* d_out, int out_size, void* d_ws, size_t ws_size,
                              hipStream_t stream) {
  const float* x     = (const float*)d_in[0];
  const float* emb   = (const float*)d_in[2];
  const float* W_ih  = (const float*)d_in[3];
  const float* W_hh  = (const float*)d_in[4];
  const float* b_ih  = (const float*)d_in[5];
  const float* b_hh  = (const float*)d_in[6];
  const float* W_out = (const float*)d_in[7];
  const float* b_out = (const float*)d_in[8];

  char* ws = (char*)d_ws;
  float*  state    = (float*)ws;   ws += (size_t)BATCH * E * 4;
  __bf16* state_b  = (__bf16*)ws;  ws += (size_t)BATCH * E * 2;
  __bf16* wout_b   = (__bf16*)ws;  ws += (size_t)V * E * 2;
  int*    last_tok = (int*)ws;     ws += (size_t)BATCH * 4;
  int*    alive    = (int*)ws;     ws += (size_t)BATCH * 4;
  int*    n_outer  = (int*)ws;     ws += (size_t)BATCH * 4;
  float*  ent_rows = (float*)ws;   ws += (size_t)BATCH * 4;
  float*  pZ       = (float*)ws;   ws += (size_t)BATCH * NSPLIT * 4;
  float*  pSL      = (float*)ws;   ws += (size_t)BATCH * NSPLIT * 4;
  float*  pBK      = (float*)ws;   ws += (size_t)BATCH * NSPLIT * 4;
  int*    pBI      = (int*)ws;     ws += (size_t)BATCH * NSPLIT * 4;

  int*   out_i = (int*)d_out;
  float* out_f = (float*)d_out;

  init_kernel<<<(BATCH * E + 255) / 256, 256, 0, stream>>>(x, state, state_b, last_tok,
                                                           alive, n_outer, ent_rows);
  convert_wout<<<(V * E + 255) / 256, 256, 0, stream>>>(W_out, wout_b);

  for (int t = 0; t < T; ++t) {
    gru_step_kernel<<<BATCH, 128, 0, stream>>>(state, state_b, last_tok, alive, emb,
                                               W_ih, W_hh, b_ih, b_hh);
    logits_partial_kernel<<<(BATCH / 128) * NSPLIT, 256, 0, stream>>>(
        state_b, wout_b, b_out, pZ, pSL, pBK, pBI, t);
    sample_finalize<<<(BATCH + 255) / 256, 256, 0, stream>>>(
        pZ, pSL, pBK, pBI, alive, last_tok, n_outer, ent_rows, out_i, t);
  }

  write_nouter<<<(BATCH + 255) / 256, 256, 0, stream>>>(n_outer, out_i);
  entropy_reduce<<<1, 256, 0, stream>>>(ent_rows, out_f);
}
